// MultiHeadAttention_27779848471522
// MI455X (gfx1250) — matrix operand order repaired and checked by reading, not compiled, not run on hardware
//
#include <hip/hip_runtime.h>
#include <hip/hip_bf16.h>

// MI455X (gfx1250) fused multi-head attention, f16 WMMA with f32 accumulation.
// B=4, S=2048, D=1024, H=16, DK=64.
// GEMMs + flash attention all run on v_wmma_f32_16x16x32_f16.
// K/V chunk staging and the output-GEMM f16 tile staging use the Tensor Data
// Mover (tensor_load_to_lds) when the builtin is available, with LDS row
// padding done by the TDM pad_interval/pad_amount fields.

typedef __attribute__((ext_vector_type(16))) _Float16 v16h;
typedef __attribute__((ext_vector_type(8)))  _Float16 v8h;
typedef __attribute__((ext_vector_type(4)))  _Float16 v4h;
typedef __attribute__((ext_vector_type(8)))  float    v8f;
typedef __attribute__((ext_vector_type(4)))  float    v4f;
typedef __attribute__((ext_vector_type(4)))  unsigned v4u;
typedef __attribute__((ext_vector_type(4)))  int      v4i;
typedef __attribute__((ext_vector_type(8)))  int      v8i;

#define MHA_B  4
#define MHA_S  2048
#define MHA_D  1024
#define MHA_H  16
#define MHA_DK 64
#define MHA_M  (MHA_B * MHA_S)   // 8192 rows for the projection GEMMs

#if defined(__gfx1250__) && __has_builtin(__builtin_amdgcn_tensor_load_to_lds) \
    && __has_builtin(__builtin_amdgcn_s_wait_tensorcnt)
#define USE_TDM 1
#else
#define USE_TDM 0
#endif

// ---------------------------------------------------------------------------
// WMMA helpers
// ---------------------------------------------------------------------------
__device__ __forceinline__ v8f wmma_f16(v16h a, v16h b, v8f c) {
    // D = A(16x32 f16) * B(32x16 f16) + C(16x16 f32)
    return __builtin_amdgcn_wmma_f32_16x16x32_f16(
        /*neg_a=*/false, a, /*neg_b=*/false, b,
        /*c_mod=*/(short)0, c, /*reuse_a=*/false, /*reuse_b=*/false);
}

// A fragment: 16(M) x 32(K) f16.  Per ISA 7.12.2:
//   lanes 0-15 (hi=0): row M=lane, K = {0..7} in v[0..7],  {16..23} in v[8..15]
//   lanes16-31 (hi=1): row M=lane-16, K = {8..15} / {24..31}
__device__ __forceinline__ v16h frag_a(const _Float16* base, int stride,
                                       int m0, int k0, int lane) {
    const int r  = lane & 15;
    const int hi = lane >> 4;
    const _Float16* p = base + (long)(m0 + r) * stride + k0 + 8 * hi;
    v8h lo = *(const v8h*)(p);
    v8h hh = *(const v8h*)(p + 16);
    return __builtin_shufflevector(lo, hh, 0, 1, 2, 3, 4, 5, 6, 7,
                                   8, 9, 10, 11, 12, 13, 14, 15);
}

// B fragment: 32(K) x 16(N) f16, stored as [N][K] row-major (K contiguous).
// lanes 0-15 hold K={0..7,16..23} (col N=lane), lanes 16-31 hold K={8..15,24..31}.
__device__ __forceinline__ v16h frag_b(const _Float16* base, int stride,
                                       int n0, int k0, int lane) {
    const int r  = lane & 15;
    const int hi = lane >> 4;
    const _Float16* p = base + (long)(n0 + r) * stride + k0 + 8 * hi;
    v8h lo = *(const v8h*)(p);
    v8h hh = *(const v8h*)(p + 16);
    return __builtin_shufflevector(lo, hh, 0, 1, 2, 3, 4, 5, 6, 7,
                                   8, 9, 10, 11, 12, 13, 14, 15);
}

#if USE_TDM
// Issue a 2D TDM load of f16 data into LDS.  Descriptor per ISA cdna5 §8:
//  group0: [1:0]=count=1, [63:32]=lds_addr, [120:64]=global_addr, [127:126]=2
//  group1: [17:16]=data_size(1=2B), [20]=pad_enable, [24:22]=pad_interval,
//          [31:25]=pad_amount, [79:48]=tensor_dim0, [111:80]=tensor_dim1,
//          [127:112]=tile_dim0, [143:128]=tile_dim1, [207:160]=dim0_stride.
// pad_interval code: dwords between pads = 2<<code; pad_amount = code+1 dwords.
// This toolchain's builtin takes 6 args (g0, g1, g2, g3, g4, cpol).
__device__ __forceinline__ void tdm_load_2d_f16(
    unsigned lds_byte, const _Float16* gptr,
    unsigned tensor_d0, unsigned tensor_d1, unsigned d0_stride,
    unsigned tile_d0, unsigned tile_d1,
    unsigned pad_interval_code, unsigned pad_amount_code)
{
    unsigned long long ga = (unsigned long long)(uintptr_t)gptr;
    v4u g0 = { 1u,                    // count = 1 valid descriptor
               lds_byte,              // LDS destination (bytes)
               (unsigned)ga,
               (unsigned)((ga >> 32) & 0x01FFFFFFull) | 0x80000000u }; // type=2
    unsigned w0 = (1u << 16)                       // data_size = 2 bytes
                | (1u << 20)                       // pad_enable
                | (pad_interval_code << 22)
                | (pad_amount_code << 25);
    v8i g1 = { (int)w0,
               (int)((tensor_d0 & 0xffffu) << 16),             // dim0 [63:48]
               (int)(((tensor_d0 >> 16) & 0xffffu)
                     | ((tensor_d1 & 0xffffu) << 16)),         // dim0 hi, dim1 lo
               (int)(((tensor_d1 >> 16) & 0xffffu)
                     | ((tile_d0 & 0xffffu) << 16)),           // dim1 hi, tile0
               (int)(tile_d1 & 0xffffu),                       // tile1 (tile2=0)
               (int)d0_stride,                                 // dim0 stride lo32
               0,                                              // stride hi, dim1 stride
               0 };
    v4i z4 = {0, 0, 0, 0};
    v8i z8 = {0, 0, 0, 0, 0, 0, 0, 0};
    __builtin_amdgcn_tensor_load_to_lds(g0, g1, z4, z4, z8, 0);
}
__device__ __forceinline__ unsigned lds_off(const void* p) {
    return (unsigned)(unsigned long long)(uintptr_t)p;   // LDS aperture low bits
}
#endif

// ---------------------------------------------------------------------------
// Projection GEMM: Y = X @ W^T + b   (X fp32 [M,K], W fp32 [N,K] -> f16 out)
// VTRANS==0 -> [B,H,S,DK] (Q, K);  VTRANS==1 -> [B,H,DK,S] (V transposed).
// Block: 256 thr (8 waves). Tile: 128(M) x 64(N), K-step 32.
// ---------------------------------------------------------------------------
template <int VTRANS>
__global__ void __launch_bounds__(256)
gemm_qkv(const float* __restrict__ X, const float* __restrict__ W,
         const float* __restrict__ bias, _Float16* __restrict__ out)
{
    __shared__ __align__(16) _Float16 As[128 * 48];  // 128 rows x 32 K (+pad)
    __shared__ __align__(16) _Float16 Ws[64 * 48];   //  64 N   x 32 K (+pad)

    const int tid  = threadIdx.x;
    const int lane = tid & 31;
    const int wave = tid >> 5;
    const int wm   = wave >> 1;      // 0..3
    const int wn   = wave & 1;       // 0..1
    const int r    = lane & 15;
    const int hi   = lane >> 4;
    const int mBlk = blockIdx.x * 128;
    const int nBlk = blockIdx.y * 64;

    v8f acc[2][2];
#pragma unroll
    for (int i = 0; i < 2; ++i)
#pragma unroll
        for (int j = 0; j < 2; ++j)
#pragma unroll
            for (int e = 0; e < 8; ++e) acc[i][j][e] = 0.0f;

    for (int k0 = 0; k0 < MHA_D; k0 += 32) {
        __syncthreads();
        // X tile: 128x32 fp32 = 1024 float4; 4 per thread. Convert to f16.
#pragma unroll
        for (int it = 0; it < 4; ++it) {
            int idx = tid + it * 256;
            int row = idx >> 3;
            int c4  = (idx & 7) * 4;
            const float* gp = X + (long)(mBlk + row) * MHA_D + k0 + c4;
            v4f xv = *(const v4f*)gp;
            if (k0 + 32 < MHA_D) __builtin_prefetch(gp + 32, 0, 0);
            v4h hv;
#pragma unroll
            for (int e = 0; e < 4; ++e) hv[e] = (_Float16)xv[e];
            *(v4h*)&As[row * 48 + c4] = hv;
        }
        // W tile: 64x32 fp32 = 512 float4; 2 per thread.
#pragma unroll
        for (int it = 0; it < 2; ++it) {
            int idx = tid + it * 256;
            int row = idx >> 3;
            int c4  = (idx & 7) * 4;
            const float* gp = W + (long)(nBlk + row) * MHA_D + k0 + c4;
            v4f wv = *(const v4f*)gp;
            if (k0 + 32 < MHA_D) __builtin_prefetch(gp + 32, 0, 0);
            v4h hv;
#pragma unroll
            for (int e = 0; e < 4; ++e) hv[e] = (_Float16)wv[e];
            *(v4h*)&Ws[row * 48 + c4] = hv;
        }
        __syncthreads();

        v16h a0 = frag_a(As, 48, wm * 32,      0, lane);
        v16h a1 = frag_a(As, 48, wm * 32 + 16, 0, lane);
        v16h b0 = frag_b(Ws, 48, wn * 32,      0, lane);
        v16h b1 = frag_b(Ws, 48, wn * 32 + 16, 0, lane);
        acc[0][0] = wmma_f16(a0, b0, acc[0][0]);
        acc[0][1] = wmma_f16(a0, b1, acc[0][1]);
        acc[1][0] = wmma_f16(a1, b0, acc[1][0]);
        acc[1][1] = wmma_f16(a1, b1, acc[1][1]);
    }

    // Epilogue: bias + head-major f16 store (layout fixed at compile time).
#pragma unroll
    for (int ti = 0; ti < 2; ++ti) {
#pragma unroll
        for (int tj = 0; tj < 2; ++tj) {
            int n  = nBlk + wn * 32 + tj * 16 + r;   // output column
            float bv = bias[n];
            int h  = n >> 6;          // head
            int dk = n & 63;          // within-head dim
#pragma unroll
            for (int i = 0; i < 8; ++i) {
                int m = mBlk + wm * 32 + ti * 16 + 8 * hi + i;  // output row
                int b = m >> 11;      // batch  (S = 2048)
                int s = m & 2047;
                float vout = acc[ti][tj][i] + bv;
                long idx;
                if (VTRANS)
                    idx = ((long)((b * MHA_H + h) * MHA_DK + dk)) * MHA_S + s;
                else
                    idx = ((long)((b * MHA_H + h) * MHA_S + s)) * MHA_DK + dk;
                out[idx] = (_Float16)vout;
            }
        }
    }
}

// ---------------------------------------------------------------------------
// Flash attention: per block, one (b,h) and 128 q rows. 8 waves x 16 q rows.
// KV chunks of 64 staged in LDS via TDM (or cooperative loads). Online softmax.
// Scale 1/sqrt(DK)=0.125 folded into Q fragments (exact power of two).
// ---------------------------------------------------------------------------
__global__ void __launch_bounds__(256)
attention(const _Float16* __restrict__ Qh, const _Float16* __restrict__ Kh,
          const _Float16* __restrict__ Vt, _Float16* __restrict__ att)
{
    __shared__ __align__(16) _Float16 Ks[64 * 72];      // [kv][dk], 144B rows
    __shared__ __align__(16) _Float16 Vs[64 * 72];      // [dk][kv]  (V^T)
    __shared__ __align__(16) _Float16 Ps[8 * 16 * 80];  // per-wave P 16x64

    const int tid  = threadIdx.x;
    const int lane = tid & 31;
    const int wave = tid >> 5;
    const int r    = lane & 15;
    const int hi   = lane >> 4;
    const int bh   = blockIdx.y;                 // b*H + h
    const int qBase = blockIdx.x * 128;

    const _Float16* Qp = Qh + (long)bh * MHA_S * MHA_DK;
    const _Float16* Kp = Kh + (long)bh * MHA_S * MHA_DK;
    const _Float16* Vp = Vt + (long)bh * MHA_DK * MHA_S;
    _Float16* Pw = &Ps[wave * 16 * 80];

    // Q fragments for this wave's 16 rows (DK=64 -> two K-steps), pre-scaled.
    v16h qa0 = frag_a(Qp, MHA_DK, qBase + wave * 16, 0,  lane);
    v16h qa1 = frag_a(Qp, MHA_DK, qBase + wave * 16, 32, lane);
#pragma unroll
    for (int j = 0; j < 16; ++j) {
        qa0[j] = qa0[j] * (_Float16)0.125f;
        qa1[j] = qa1[j] * (_Float16)0.125f;
    }

    v8f oc[4];
#pragma unroll
    for (int t = 0; t < 4; ++t)
#pragma unroll
        for (int e = 0; e < 8; ++e) oc[t][e] = 0.0f;

    float mrun[8], lrun[8];
#pragma unroll
    for (int i = 0; i < 8; ++i) { mrun[i] = -3.0e38f; lrun[i] = 0.0f; }

    for (int kv0 = 0; kv0 < MHA_S; kv0 += 64) {
        __syncthreads();
#if USE_TDM
        // Tensor Data Mover: stage K[kv0..+63][0..63] and V^T[0..63][kv0..+63].
        // Tile rows: 64 f16 = 128B = 32 dwords -> pad_interval code 4 (32 dw);
        // pad 16B = 4 dwords -> code 3 => LDS row stride 144B (= 72 halfs).
        if (wave == 0) {
            tdm_load_2d_f16(lds_off(Ks), Kp + (long)kv0 * MHA_DK,
                            /*t_d0=*/MHA_DK, /*t_d1=*/MHA_S, /*stride=*/MHA_DK,
                            /*tile0=*/64, /*tile1=*/64, /*padi=*/4, /*pada=*/3);
            tdm_load_2d_f16(lds_off(Vs), Vp + kv0,
                            /*t_d0=*/MHA_S, /*t_d1=*/MHA_DK, /*stride=*/MHA_S,
                            /*tile0=*/64, /*tile1=*/64, /*padi=*/4, /*pada=*/3);
            __builtin_amdgcn_s_wait_tensorcnt(0);
        }
#else
        // Cooperative staging: 512 v8h each; 2 per thread each.
#pragma unroll
        for (int it = 0; it < 2; ++it) {
            int idx = tid + it * 256;
            int row = idx >> 3;
            int c8  = (idx & 7) * 8;
            *(v8h*)&Ks[row * 72 + c8] =
                *(const v8h*)(Kp + (long)(kv0 + row) * MHA_DK + c8);
            *(v8h*)&Vs[row * 72 + c8] =
                *(const v8h*)(Vp + (long)row * MHA_S + kv0 + c8);
        }
#endif
        __syncthreads();

        // Scores: S = (Q*0.125) @ K^T -> 16 x 64 (4 N-tiles, 2 K-steps each)
        v8f sc[4];
#pragma unroll
        for (int t = 0; t < 4; ++t) {
            v8f z;
#pragma unroll
            for (int e = 0; e < 8; ++e) z[e] = 0.0f;
            sc[t] = wmma_f16(qa0, frag_b(Ks, 72, 16 * t, 0,  lane), z);
            sc[t] = wmma_f16(qa1, frag_b(Ks, 72, 16 * t, 32, lane), sc[t]);
        }

        // Online softmax. C layout: element e of lane = (row e+8*hi, col r);
        // row reductions stay within a 16-lane half (xor masks 1,2,4,8).
        float mnew[8], fsc[8], lch[8];
#pragma unroll
        for (int i = 0; i < 8; ++i) {
            float mx = sc[0][i];
#pragma unroll
            for (int t = 1; t < 4; ++t) mx = fmaxf(mx, sc[t][i]);
#pragma unroll
            for (int d = 1; d < 16; d <<= 1)
                mx = fmaxf(mx, __shfl_xor(mx, d, 32));
            mnew[i] = fmaxf(mrun[i], mx);
            fsc[i]  = __expf(mrun[i] - mnew[i]);
            mrun[i] = mnew[i];
            lch[i]  = 0.0f;
        }
#pragma unroll
        for (int t = 0; t < 4; ++t) {
#pragma unroll
            for (int i = 0; i < 8; ++i) {
                float p = __expf(sc[t][i] - mnew[i]);
                lch[i] += p;
                Pw[(i + 8 * hi) * 80 + 16 * t + r] = (_Float16)p;
            }
        }
#pragma unroll
        for (int i = 0; i < 8; ++i) {
#pragma unroll
            for (int d = 1; d < 16; d <<= 1)
                lch[i] += __shfl_xor(lch[i], d, 32);
            lrun[i] = lrun[i] * fsc[i] + lch[i];
        }
        // Rescale running output accumulators.
#pragma unroll
        for (int t = 0; t < 4; ++t)
#pragma unroll
            for (int i = 0; i < 8; ++i) oc[t][i] *= fsc[i];

        // O += P @ V  (P 16x64 from per-wave LDS; V^T B-frags from Vs)
        v16h pa0 = frag_a(Pw, 80, 0, 0,  lane);
        v16h pa1 = frag_a(Pw, 80, 0, 32, lane);
#pragma unroll
        for (int t = 0; t < 4; ++t) {
            oc[t] = wmma_f16(pa0, frag_b(Vs, 72, 16 * t, 0,  lane), oc[t]);
            oc[t] = wmma_f16(pa1, frag_b(Vs, 72, 16 * t, 32, lane), oc[t]);
        }
    }

    // Epilogue: normalize by row sums, store f16 into [B,S,D] concat layout.
    const int b = bh >> 4;
    const int h = bh & 15;
#pragma unroll
    for (int t = 0; t < 4; ++t) {
#pragma unroll
        for (int i = 0; i < 8; ++i) {
            int qs = qBase + wave * 16 + 8 * hi + i;
            int dc = h * MHA_DK + 16 * t + r;
            att[((long)b * MHA_S + qs) * MHA_D + dc] =
                (_Float16)(oc[t][i] / lrun[i]);
        }
    }
}

// ---------------------------------------------------------------------------
// Output GEMM: out = att(f16) @ Wo^T + bo -> fp32 [M, D]
// X tile staged by TDM (f16, no conversion needed); W tile converted by VALU.
// ---------------------------------------------------------------------------
__global__ void __launch_bounds__(256)
gemm_out(const _Float16* __restrict__ X, const float* __restrict__ W,
         const float* __restrict__ bias, float* __restrict__ out)
{
    __shared__ __align__(16) _Float16 As[128 * 48];  // 96B rows
    __shared__ __align__(16) _Float16 Ws[64 * 48];

    const int tid  = threadIdx.x;
    const int lane = tid & 31;
    const int wave = tid >> 5;
    const int wm   = wave >> 1;
    const int wn   = wave & 1;
    const int r    = lane & 15;
    const int hi   = lane >> 4;
    const int mBlk = blockIdx.x * 128;
    const int nBlk = blockIdx.y * 64;

    v8f acc[2][2];
#pragma unroll
    for (int i = 0; i < 2; ++i)
#pragma unroll
        for (int j = 0; j < 2; ++j)
#pragma unroll
            for (int e = 0; e < 8; ++e) acc[i][j][e] = 0.0f;

    for (int k0 = 0; k0 < MHA_D; k0 += 32) {
        __syncthreads();
#if USE_TDM
        // X tile 128x32 f16: rows 32 halfs = 64B = 16 dwords -> interval code 3;
        // pad 32B = 8 dwords -> code 7 => LDS row stride 96B (= 48 halfs).
        if (wave == 0) {
            tdm_load_2d_f16(lds_off(As), X + (long)mBlk * MHA_D + k0,
                            /*t_d0=*/MHA_D, /*t_d1=*/MHA_M, /*stride=*/MHA_D,
                            /*tile0=*/32, /*tile1=*/128, /*padi=*/3, /*pada=*/7);
        }
#else
#pragma unroll
        for (int it = 0; it < 2; ++it) {
            int idx = tid + it * 256;
            int row = idx >> 2;
            int c8  = (idx & 3) * 8;
            *(v8h*)&As[row * 48 + c8] =
                *(const v8h*)(X + (long)(mBlk + row) * MHA_D + k0 + c8);
        }
#endif
        // W tile: 64x32 fp32 -> f16.
#pragma unroll
        for (int it = 0; it < 2; ++it) {
            int idx = tid + it * 256;
            int row = idx >> 3;
            int c4  = (idx & 7) * 4;
            const float* gp = W + (long)(nBlk + row) * MHA_D + k0 + c4;
            v4f wv = *(const v4f*)gp;
            if (k0 + 32 < MHA_D) __builtin_prefetch(gp + 32, 0, 0);
            v4h hv;
#pragma unroll
            for (int e = 0; e < 4; ++e) hv[e] = (_Float16)wv[e];
            *(v4h*)&Ws[row * 48 + c4] = hv;
        }
#if USE_TDM
        if (wave == 0) __builtin_amdgcn_s_wait_tensorcnt(0);
#endif
        __syncthreads();

        v16h a0 = frag_a(As, 48, wm * 32,      0, lane);
        v16h a1 = frag_a(As, 48, wm * 32 + 16, 0, lane);
        v16h b0 = frag_b(Ws, 48, wn * 32,      0, lane);
        v16h b1 = frag_b(Ws, 48, wn * 32 + 16, 0, lane);
        acc[0][0] = wmma_f16(a0, b0, acc[0][0]);
        acc[0][1] = wmma_f16(a0, b1, acc[0][1]);
        acc[1][0] = wmma_f16(a1, b0, acc[1][0]);
        acc[1][1] = wmma_f16(a1, b1, acc[1][1]);
    }

#pragma unroll
    for (int ti = 0; ti < 2; ++ti) {
#pragma unroll
        for (int tj = 0; tj < 2; ++tj) {
            int n = nBlk + wn * 32 + tj * 16 + r;
            float bv = bias[n];
#pragma unroll
            for (int i = 0; i < 8; ++i) {
                int m = mBlk + wm * 32 + ti * 16 + 8 * hi + i;
                out[(long)m * MHA_D + n] = acc[ti][tj][i] + bv;
            }
        }
    }
}

// ---------------------------------------------------------------------------
// Launcher. Inputs: q,k,v,Wq,bq,Wk,bk,Wv,bv,Wo,bo (fp32). Output fp32 [B,S,D].
// Workspace: Qh(16MB) Kh(16MB) V^T(16MB) att(16MB) = 64MB of f16 buffers.
// ---------------------------------------------------------------------------
extern "C" void kernel_launch(void* const* d_in, const int* in_sizes, int n_in,
                              void* d_out, int out_size, void* d_ws, size_t ws_size,
                              hipStream_t stream)
{
    (void)in_sizes; (void)n_in; (void)out_size; (void)ws_size;

    const float* q  = (const float*)d_in[0];
    const float* k  = (const float*)d_in[1];
    const float* v  = (const float*)d_in[2];
    const float* Wq = (const float*)d_in[3];
    const float* bq = (const float*)d_in[4];
    const float* Wk = (const float*)d_in[5];
    const float* bk = (const float*)d_in[6];
    const float* Wv = (const float*)d_in[7];
    const float* bv = (const float*)d_in[8];
    const float* Wo = (const float*)d_in[9];
    const float* bo = (const float*)d_in[10];
    float* out = (float*)d_out;

    char* ws = (char*)d_ws;
    const size_t BUF = (size_t)MHA_B * MHA_S * MHA_D * sizeof(_Float16); // 16MB
    _Float16* Qh  = (_Float16*)(ws + 0 * BUF);
    _Float16* Kh  = (_Float16*)(ws + 1 * BUF);
    _Float16* Vt  = (_Float16*)(ws + 2 * BUF);
    _Float16* att = (_Float16*)(ws + 3 * BUF);

    dim3 blk(256);
    dim3 gGemm(MHA_M / 128, MHA_D / 64);      // (64, 16)
    dim3 gAttn(MHA_S / 128, MHA_B * MHA_H);   // (16, 64)

    gemm_qkv<0><<<gGemm, blk, 0, stream>>>(q, Wq, bq, Qh);
    gemm_qkv<0><<<gGemm, blk, 0, stream>>>(k, Wk, bk, Kh);
    gemm_qkv<1><<<gGemm, blk, 0, stream>>>(v, Wv, bv, Vt);
    attention<<<gAttn, blk, 0, stream>>>(Qh, Kh, Vt, att);
    gemm_out<<<gGemm, blk, 0, stream>>>(att, Wo, bo, out);
}
